// GAT_57080115364090
// MI455X (gfx1250) — matrix operand order repaired and checked by reading, not compiled, not run on hardware
//
#include <hip/hip_runtime.h>
#include <hip/hip_bf16.h>

// ---------------------------------------------------------------------------
// GATv2Conv forward for MI455X (gfx1250, wave32).
// GEMM via v_wmma_f32_16x16x32_bf16 with 4x2 register blocking per wave;
// edge phases are L2-resident gathers/scatters.
// ---------------------------------------------------------------------------

typedef __attribute__((ext_vector_type(16))) __bf16 v16bf;   // 8 VGPRs
typedef __attribute__((ext_vector_type(8)))  __bf16 v8bf;    // 4 VGPRs (16B)
typedef __attribute__((ext_vector_type(8)))  float  v8f;     // 8 VGPRs

#define DD   512      // feature dim
#define DD2  1024     // concat(W_l, W_r) output dim
#define NH   8        // heads
#define CC   64       // channels / head

// ---------------------------------------------------------------------------
// prep: x -> bf16, build Wcat^T in bf16 (column-major so B fragments are
// contiguous), zero output accumulator / segment-max keys / denominators.
// ---------------------------------------------------------------------------
__global__ void gat_prep_kernel(const float* __restrict__ x,
                                const float* __restrict__ wl,
                                const float* __restrict__ wr,
                                __bf16* __restrict__ xb,
                                __bf16* __restrict__ wt,      // [1024][512]
                                float* __restrict__ acc,      // d_out, zeroed
                                unsigned* __restrict__ amaxk, // [N*H]
                                float* __restrict__ denom,    // [N*H]
                                int nd, int nh)
{
    int tid = blockIdx.x * blockDim.x + threadIdx.x;
    if (tid < nd) {
        xb[tid]  = (__bf16)x[tid];
        acc[tid] = 0.0f;
    }
    if (tid < DD2 * DD) {
        int n = tid >> 9;          // output column 0..1023
        int k = tid & (DD - 1);    // input dim 0..511
        float v = (n < DD) ? wl[(size_t)k * DD + n]
                           : wr[(size_t)k * DD + (n - DD)];
        wt[tid] = (__bf16)v;
    }
    if (tid < nh) {
        amaxk[tid] = 0u;           // key-space minimum (below key(-inf))
        denom[tid] = 0.0f;
    }
}

// ---------------------------------------------------------------------------
// GEMM: xlxr[N,1024] = bf16(x)[N,512] @ bf16(Wcat)[512,1024], f32 accumulate.
// Each wave computes a 64x32 patch: 4 m-tiles x 2 n-tiles (8 accumulators),
// K in steps of 32 via v_wmma_f32_16x16x32_bf16.
// Per K-step: 4 A fragments + 2 B fragments -> 8 WMMAs (~21 FLOP/L2-byte).
// Fragment K order is contiguous (k = 16*sel + s), the same in A and B.
// ---------------------------------------------------------------------------
__global__ __launch_bounds__(256) void gat_gemm_kernel(
    const __bf16* __restrict__ xb,    // [N,512] row-major
    const __bf16* __restrict__ wt,    // [1024,512] (Wcat^T row-major)
    float* __restrict__ xlxr,         // [N,1024]
    int nrows)
{
    const int lane = threadIdx.x & 31;
    const int wave = threadIdx.x >> 5;
    const int m0   = blockIdx.x * 64;                // 4 m-tiles per wave
    const int n0   = (blockIdx.y * 8 + wave) * 32;   // 2 n-tiles per wave

    const int sel = lane >> 4;   // half-wave select
    const int r   = lane & 15;

    // A: lane holds row m0+mt*16+r; v0..7 = K[16*sel..+15] (contiguous)
    const __bf16* ap[4];
#pragma unroll
    for (int mt = 0; mt < 4; ++mt) {
        int arow = min(m0 + mt * 16 + r, nrows - 1);  // clamp keeps EXEC full
        ap[mt] = xb + (size_t)arow * DD + sel * 16;
    }
    // B: lane holds column n0(+16)+r of Wcat; 16 contiguous K at kb + 16*sel
    const __bf16* bp0 = wt + (size_t)(n0 + r) * DD + sel * 16;
    const __bf16* bp1 = wt + (size_t)(n0 + 16 + r) * DD + sel * 16;

    v8f c[4][2] = {};
#pragma unroll 2
    for (int kb = 0; kb < DD; kb += 32) {
        v16bf b0 = *(const v16bf*)(bp0 + kb);
        v16bf b1 = *(const v16bf*)(bp1 + kb);
#pragma unroll
        for (int mt = 0; mt < 4; ++mt) {
            v16bf a;
            ((v8bf*)&a)[0] = *(const v8bf*)(ap[mt] + kb);
            ((v8bf*)&a)[1] = *(const v8bf*)(ap[mt] + kb + 8);
            c[mt][0] = __builtin_amdgcn_wmma_f32_16x16x32_bf16(
                           false, a, false, b0, (short)0, c[mt][0], false, false);
            c[mt][1] = __builtin_amdgcn_wmma_f32_16x16x32_bf16(
                           false, a, false, b1, (short)0, c[mt][1], false, false);
        }
    }

    // D layout: VGPR i -> row base + 8*sel + i, col n0(+16) + (lane&15)
#pragma unroll
    for (int mt = 0; mt < 4; ++mt) {
        const int rowbase = m0 + mt * 16 + sel * 8;
        float* out = xlxr + (size_t)rowbase * DD2 + n0 + r;
#pragma unroll
        for (int i = 0; i < 8; ++i) {
            if (rowbase + i < nrows) {
                out[(size_t)i * DD2]      = c[mt][0][i];
                out[(size_t)i * DD2 + 16] = c[mt][1][i];
            }
        }
    }
}

// ---------------------------------------------------------------------------
// Edge logits: one wave per edge. Lane l owns channels [16*l, 16*l+16)
// (att flat index == 16*l). Reduce groups of 4 lanes -> 8 head logits.
// Segment max via order-preserving uint key atomicMax.
// ---------------------------------------------------------------------------
__device__ __forceinline__ unsigned f32_key(float x) {
    unsigned b = __float_as_uint(x);
    return (b & 0x80000000u) ? ~b : (b | 0x80000000u);
}
__device__ __forceinline__ float key_f32(unsigned k) {
    unsigned b = (k & 0x80000000u) ? (k & 0x7FFFFFFFu) : ~k;
    return __uint_as_float(b);
}

__global__ __launch_bounds__(256) void gat_logits_kernel(
    const int* __restrict__ ei,       // [2,E]
    const float* __restrict__ xlxr,   // [N,1024]  (x_l | x_r)
    const float* __restrict__ att,    // [512]
    float* __restrict__ alphab,       // [(E+N)*8]
    unsigned* __restrict__ amaxk,     // [N*8]
    int E_, int ET)
{
    const int lane = threadIdx.x & 31;
    const int e = blockIdx.x * 8 + (threadIdx.x >> 5);
    if (e >= ET) return;
    int src, dst;
    if (e < E_) { src = ei[e]; dst = ei[E_ + e]; }
    else        { src = dst = e - E_; }

    const float4* pl = (const float4*)(xlxr + (size_t)src * DD2 + lane * 16);
    const float4* pr = (const float4*)(xlxr + (size_t)dst * DD2 + DD + lane * 16);
    const float4* pa = (const float4*)(att + lane * 16);

    float s = 0.0f;
#pragma unroll
    for (int i = 0; i < 4; ++i) {
        float4 a = pl[i], b = pr[i], w = pa[i];
        float v;
        v = a.x + b.x; v = v > 0.f ? v : 0.2f * v; s += w.x * v;
        v = a.y + b.y; v = v > 0.f ? v : 0.2f * v; s += w.y * v;
        v = a.z + b.z; v = v > 0.f ? v : 0.2f * v; s += w.z * v;
        v = a.w + b.w; v = v > 0.f ? v : 0.2f * v; s += w.w * v;
    }
    s += __shfl_xor(s, 1, 32);
    s += __shfl_xor(s, 2, 32);
    if ((lane & 3) == 0) {
        int h = lane >> 2;
        alphab[(size_t)e * NH + h] = s;
        atomicMax(amaxk + (size_t)dst * NH + h, f32_key(s));
    }
}

// ---------------------------------------------------------------------------
// exp(alpha - segmax) and denominator accumulation. Thread per (edge, head).
// ---------------------------------------------------------------------------
__global__ void gat_exp_kernel(const int* __restrict__ ei,
                               float* __restrict__ alphab,
                               const unsigned* __restrict__ amaxk,
                               float* __restrict__ denom,
                               int E_, int ET)
{
    int tid = blockIdx.x * blockDim.x + threadIdx.x;
    if (tid >= ET * NH) return;
    int e = tid >> 3, h = tid & 7;
    int dst = (e < E_) ? ei[E_ + e] : (e - E_);
    float m  = key_f32(amaxk[(size_t)dst * NH + h]);
    float ea = __expf(alphab[tid] - m);
    alphab[tid] = ea;                      // reuse buffer for exp(alpha)
    atomicAdd(denom + (size_t)dst * NH + h, ea);
}

// ---------------------------------------------------------------------------
// Weighted scatter-aggregate: one wave per edge, 16 f32 atomics per lane.
// ---------------------------------------------------------------------------
__global__ __launch_bounds__(256) void gat_aggr_kernel(
    const int* __restrict__ ei,
    const float* __restrict__ xlxr,
    const float* __restrict__ alphab,
    const float* __restrict__ denom,
    float* __restrict__ acc,            // [N,512] (in d_out)
    int E_, int ET)
{
    const int lane = threadIdx.x & 31;
    const int e = blockIdx.x * 8 + (threadIdx.x >> 5);
    if (e >= ET) return;
    int src, dst;
    if (e < E_) { src = ei[e]; dst = ei[E_ + e]; }
    else        { src = dst = e - E_; }

    const int h = lane >> 2;
    float d = denom[(size_t)dst * NH + h];
    float w = alphab[(size_t)e * NH + h] / fmaxf(d, 1e-16f);

    const float4* pl = (const float4*)(xlxr + (size_t)src * DD2 + lane * 16);
    float* po = acc + (size_t)dst * DD + lane * 16;
#pragma unroll
    for (int i = 0; i < 4; ++i) {
        float4 a = pl[i];
        atomicAdd(po + 4 * i + 0, w * a.x);
        atomicAdd(po + 4 * i + 1, w * a.y);
        atomicAdd(po + 4 * i + 2, w * a.z);
        atomicAdd(po + 4 * i + 3, w * a.w);
    }
}

// ---------------------------------------------------------------------------
// finalize: out = elu(acc + bias) + x   (in place in d_out)
// ---------------------------------------------------------------------------
__global__ void gat_final_kernel(const float* __restrict__ x,
                                 const float* __restrict__ bias,
                                 float* __restrict__ out, int total)
{
    int tid = blockIdx.x * blockDim.x + threadIdx.x;
    if (tid >= total) return;
    float v = out[tid] + bias[tid & (DD - 1)];
    v = v > 0.0f ? v : expm1f(v);
    out[tid] = v + x[tid];
}

// ---------------------------------------------------------------------------
extern "C" void kernel_launch(void* const* d_in, const int* in_sizes, int n_in,
                              void* d_out, int out_size, void* d_ws, size_t ws_size,
                              hipStream_t stream)
{
    const float* x    = (const float*)d_in[0];
    const int*   ei   = (const int*)  d_in[1];
    const float* wl   = (const float*)d_in[2];
    const float* wr   = (const float*)d_in[3];
    const float* att  = (const float*)d_in[4];
    const float* bias = (const float*)d_in[5];
    float* out = (float*)d_out;

    const int N_ = in_sizes[0] / DD;       // 10000
    const int E_ = in_sizes[1] / 2;        // 160000
    const int ET = E_ + N_;                // 170000 (self loops appended)
    const int ND = N_ * DD;                // 5,120,000
    const int NHn = N_ * NH;

    // workspace layout (256B-aligned partitions)
    char* ws = (char*)d_ws;
    size_t off = 0;
    __bf16*  xb     = (__bf16*)(ws + off);  off += (size_t)ND * 2;              // 10.24 MB
    __bf16*  wt     = (__bf16*)(ws + off);  off += (size_t)DD2 * DD * 2;        //  1.05 MB
    float*   xlxr   = (float*)(ws + off);   off += (size_t)N_ * DD2 * 4;        // 40.96 MB
    float*   alphab = (float*)(ws + off);   off += (size_t)ET * NH * 4;         //  5.44 MB
    unsigned* amaxk = (unsigned*)(ws + off);off += (size_t)NHn * 4;             //  0.32 MB
    float*   denom  = (float*)(ws + off);   off += (size_t)NHn * 4;             //  0.32 MB

    // 1) convert + transpose + zero-init (must run every call: ws/out are dirty)
    gat_prep_kernel<<<(ND + 255) / 256, 256, 0, stream>>>(
        x, wl, wr, xb, wt, out, amaxk, denom, ND, NHn);

    // 2) WMMA GEMM: [N,512] x [512,1024] -> [N,1024]
    //    block = 8 waves, each wave a 64x32 patch -> block covers 64x256
    dim3 ggrid((N_ + 63) / 64, DD2 / 256);   // 157 x 4
    gat_gemm_kernel<<<ggrid, 256, 0, stream>>>(xb, wt, xlxr, N_);

    // 3) attention logits + segment max
    int eblocks = (ET + 7) / 8;
    gat_logits_kernel<<<eblocks, 256, 0, stream>>>(ei, xlxr, att, alphab, amaxk, E_, ET);

    // 4) exp + denominators
    gat_exp_kernel<<<(ET * NH + 255) / 256, 256, 0, stream>>>(ei, alphab, amaxk, denom, E_, ET);

    // 5) weighted scatter aggregate into d_out
    gat_aggr_kernel<<<eblocks, 256, 0, stream>>>(ei, xlxr, alphab, denom, out, E_, ET);

    // 6) elu + bias + residual (in place)
    gat_final_kernel<<<(ND + 255) / 256, 256, 0, stream>>>(x, bias, out, ND);
}
